// NN_91268055040275
// MI455X (gfx1250) — matrix operand order repaired and checked by reading, not compiled, not run on hardware
//
#include <hip/hip_runtime.h>

// ---------------------------------------------------------------------------
// MI455X (gfx1250, wave32) fused feature + 3-layer LN/tanh MLP using
// v_wmma_f32_16x16x32_f16. Compute-bound workload (~284 GFLOP, only ~85MB
// HBM traffic), so everything rides the WMMA pipe in f16 with f32 accum.
// B-fragment loads are batched per k-step ahead of the WMMA block, and the
// wave-private weight stream is prefetched at WGP scope (inline asm -- the
// __builtin_prefetch lowering uses SCOPE_SE, which skips the WGP cache).
// ---------------------------------------------------------------------------

typedef __attribute__((ext_vector_type(16))) _Float16 v16h;
typedef __attribute__((ext_vector_type(8)))  _Float16 v8h;
typedef __attribute__((ext_vector_type(8)))  float    v8f;

#define STR      1032   // LDS row stride in halves: 2064B rows -> lanes hit
                        // distinct bank groups for b128 A-fragment loads
#define ROWS     32     // batch rows per workgroup
#define NTHREADS 512    // 16 wave32s

__device__ __forceinline__ float fast_tanh(float x) {
  // tanh(x) = 1 - 2/(exp(2x)+1); v_exp_f32 + v_rcp_f32, saturates correctly.
  float e = __builtin_amdgcn_exp2f(x * 2.8853900817779268f); // 2*log2(e)
  return 1.0f - 2.0f * __builtin_amdgcn_rcpf(e + 1.0f);
}

// WGP-scope prefetch (scope field omitted -> 0 = WGP: fill all cache levels).
__device__ __forceinline__ void prefetch_wgp(const void* p) {
  asm volatile("global_prefetch_b8 %0, off offset:256" :: "v"(p));
}

// A-fragment (16x32 f16): lane l holds row M=l&15; K contiguous
// [k0 + (l>>4)*16 .. +15] as two 16B LDS loads.
__device__ __forceinline__ v16h load_a(const _Float16* p) {
  v8h lo = *(const v8h*)p;
  v8h hi = *(const v8h*)(p + 8);
  return __builtin_shufflevector(lo, hi, 0,1,2,3,4,5,6,7,8,9,10,11,12,13,14,15);
}

// One layer: out = tanh(LN(in @ W^T + b) * g + be), M=32 rows, K->N.
// Wave `wid` owns both 16-row tiles and T=N/256 column tiles (B frag reused
// across the two A tiles -> each weight element read from L2 once per WG).
template<int K, int N>
__device__ __forceinline__ void layer(const _Float16* __restrict__ in,
                                      _Float16*       __restrict__ out,
                                      const _Float16* __restrict__ W,
                                      const float*    __restrict__ bias,
                                      const float*    __restrict__ g,
                                      const float*    __restrict__ be,
                                      float* __restrict__ s_sum,
                                      float* __restrict__ s_sq,
                                      int wid, int lane)
{
  constexpr int T = N / 256;         // 16x16 col-tiles per wave
  const int l16 = lane & 15;
  const int lh  = lane >> 4;

  if (threadIdx.x < ROWS) { s_sum[threadIdx.x] = 0.f; s_sq[threadIdx.x] = 0.f; }
  __syncthreads();                   // stats zeroed + `in` visible

  // Per-tile weight stream base: lane l -> column N = n0 + (l&15),
  // 16 contiguous halves at K = k0 + (l>>4)*16 (32B-aligned global loads).
  const _Float16* wbase[T];
#pragma unroll
  for (int t = 0; t < T; ++t)
    wbase[t] = W + (size_t)((wid * T + t) * 16 + l16) * K + lh * 16;

  v8f acc[2][T] = {};

  for (int k0 = 0; k0 < K; k0 += 32) {
    // batch all fragment loads for this k-step first
    v16h a0 = load_a(in + (l16)      * STR + k0 + lh * 16);
    v16h a1 = load_a(in + (16 + l16) * STR + k0 + lh * 16);
    v16h b[T];
#pragma unroll
    for (int t = 0; t < T; ++t) {
      b[t] = *(const v16h*)(wbase[t] + k0);
      prefetch_wgp(wbase[t] + k0);   // +256B ahead, WGP scope
    }
    // then the 2*T-WMMA block (B frag reused across both row tiles)
#pragma unroll
    for (int t = 0; t < T; ++t) {
      acc[0][t] = __builtin_amdgcn_wmma_f32_16x16x32_f16(
          false, a0, false, b[t], (short)0, acc[0][t], false, false);
      acc[1][t] = __builtin_amdgcn_wmma_f32_16x16x32_f16(
          false, a1, false, b[t], (short)0, acc[1][t], false, false);
    }
  }

  // bias add (per output column)
#pragma unroll
  for (int t = 0; t < T; ++t) {
    float bv = bias[(wid * T + t) * 16 + l16];
#pragma unroll
    for (int rt = 0; rt < 2; ++rt)
#pragma unroll
      for (int j = 0; j < 8; ++j) acc[rt][t][j] += bv;
  }

  // LayerNorm stats: C layout -> VGPR j is row M = rt*16 + j + lh*8, N = l16.
#pragma unroll
  for (int rt = 0; rt < 2; ++rt)
#pragma unroll
    for (int j = 0; j < 8; ++j) {
      float sv = 0.f, qv = 0.f;
#pragma unroll
      for (int t = 0; t < T; ++t) { float v = acc[rt][t][j]; sv += v; qv += v * v; }
#pragma unroll
      for (int m = 1; m < 16; m <<= 1) {       // reduce within 16-lane halves
        sv += __shfl_xor(sv, m, 32);
        qv += __shfl_xor(qv, m, 32);
      }
      if (l16 == 0) {
        int row = rt * 16 + j + lh * 8;
        atomicAdd(&s_sum[row], sv);            // ds_add_f32
        atomicAdd(&s_sq[row],  qv);
      }
    }
  __syncthreads();                             // stats complete

  const float invN = 1.0f / (float)N;
#pragma unroll
  for (int t = 0; t < T; ++t) {
    int n = (wid * T + t) * 16 + l16;
    float gv = g[n], bev = be[n];
#pragma unroll
    for (int rt = 0; rt < 2; ++rt)
#pragma unroll
      for (int j = 0; j < 8; ++j) {
        int row = rt * 16 + j + lh * 8;
        float mean = s_sum[row] * invN;
        float var  = s_sq[row] * invN - mean * mean;
        float rs   = __builtin_amdgcn_rsqf(var + 1e-5f);
        float v    = (acc[rt][t][j] - mean) * rs * gv + bev;
        out[row * STR + n] = (_Float16)fast_tanh(v);
      }
  }
  __syncthreads();                             // out ready; stats free to reuse
}

__global__ __launch_bounds__(NTHREADS) void mlp_fused(
    const float* __restrict__ x,
    const float* __restrict__ kc,  const float* __restrict__ nl,
    const float* __restrict__ opw, const float* __restrict__ opb,
    const float* __restrict__ ic,  const float* __restrict__ isb,
    const float* __restrict__ ii,
    const _Float16* __restrict__ W0h, const float* __restrict__ b0,
    const _Float16* __restrict__ W1h, const float* __restrict__ b1,
    const _Float16* __restrict__ W2h, const float* __restrict__ b2,
    const float* __restrict__ W3,     const float* __restrict__ b3,
    const float* __restrict__ g0, const float* __restrict__ be0,
    const float* __restrict__ g1, const float* __restrict__ be1,
    const float* __restrict__ g2, const float* __restrict__ be2,
    float* __restrict__ outp)
{
  extern __shared__ __align__(16) char smem[];
  _Float16* bufA = (_Float16*)smem;            // 32 x STR halves
  _Float16* bufB = bufA + ROWS * STR;          // 32 x STR halves
  float* s_sum = (float*)(bufB + ROWS * STR);  // 32 f32
  float* s_sq  = s_sum + ROWS;                 // 32 f32

  const int tid  = threadIdx.x;
  const int wid  = tid >> 5;
  const int lane = tid & 31;
  const int row0 = blockIdx.x * ROWS;

  // ---- feature build: 16 threads per row, 36 features each, f16 into bufA
  {
    int row  = tid >> 4;                       // 0..31
    int gsub = tid & 15;
    const float* xr = x + (size_t)(row0 + row) * 323;
    for (int j = gsub; j < 576; j += 16) {
      float v;
      if (j < 32)       v = kc[((int)xr[0]) * 32 + j];
      else if (j < 64)  v = nl[((int)xr[1]) * 32 + (j - 32)];
      else if (j < 176) { int jj = j - 64; int k = jj >> 1, c = jj & 1;
                          v = xr[2 + k] * opw[k * 2 + c] + opb[k * 2 + c]; }
      else if (j < 178) v = xr[58 + (j - 176)];
      else if (j < 568) {
        int jj = j - 178; int bl = jj / 26, o = jj - bl * 26;
        const float* xb = xr + 60 + bl * 17;
        if (o < 14)      v = xb[o];
        else if (o < 18) v = ic [((int)xb[14]) * 4 + (o - 14)];
        else if (o < 22) v = isb[((int)xb[15]) * 4 + (o - 18)];
        else             v = ii [((int)xb[16]) * 4 + (o - 22)];
      } else            v = xr[315 + (j - 568)];
      bufA[row * STR + j] = (_Float16)v;
    }
  }
  // layer<>'s leading __syncthreads orders feature writes before GEMM reads.

  layer< 576, 1024>(bufA, bufB, W0h, b0, g0, be0, s_sum, s_sq, wid, lane);
  layer<1024, 1024>(bufB, bufA, W1h, b1, g1, be1, s_sum, s_sq, wid, lane);
  layer<1024,  512>(bufA, bufB, W2h, b2, g2, be2, s_sum, s_sq, wid, lane);

  // ---- final 512 -> 1 dot product (h2 in bufB)
  {
    int row = tid >> 4, l = tid & 15;
    float s = 0.f;
    for (int k = l; k < 512; k += 16)
      s += (float)bufB[row * STR + k] * W3[k];
#pragma unroll
    for (int m = 1; m < 16; m <<= 1) s += __shfl_xor(s, m, 32);
    if (l == 0) outp[row0 + row] = s + b3[0];
  }
}

// f32 -> f16 weight conversion into workspace (weights then live in L2).
__global__ void convert_weights(const float* __restrict__ W0,
                                const float* __restrict__ W1,
                                const float* __restrict__ W2,
                                _Float16* __restrict__ wsh)
{
  size_t i = (size_t)blockIdx.x * blockDim.x + threadIdx.x;
  if (i <  589824u) wsh[i] = (_Float16)W0[i];                       // 1024x576
  if (i < 1048576u) wsh[589824u + i] = (_Float16)W1[i];             // 1024x1024
  if (i <  524288u) wsh[589824u + 1048576u + i] = (_Float16)W2[i];  // 512x1024
}

extern "C" void kernel_launch(void* const* d_in, const int* in_sizes, int n_in,
                              void* d_out, int out_size, void* d_ws, size_t ws_size,
                              hipStream_t stream)
{
  const float* x   = (const float*)d_in[0];
  const float* kc  = (const float*)d_in[1];
  const float* nl  = (const float*)d_in[2];
  const float* opw = (const float*)d_in[3];
  const float* opb = (const float*)d_in[4];
  const float* ic  = (const float*)d_in[5];
  const float* isb = (const float*)d_in[6];
  const float* ii  = (const float*)d_in[7];
  const float* W0  = (const float*)d_in[8];
  const float* b0  = (const float*)d_in[9];
  const float* W1  = (const float*)d_in[10];
  const float* b1  = (const float*)d_in[11];
  const float* W2  = (const float*)d_in[12];
  const float* b2  = (const float*)d_in[13];
  const float* W3  = (const float*)d_in[14];
  const float* b3  = (const float*)d_in[15];
  const float* g0  = (const float*)d_in[16];
  const float* be0 = (const float*)d_in[17];
  const float* g1  = (const float*)d_in[18];
  const float* be1 = (const float*)d_in[19];
  const float* g2  = (const float*)d_in[20];
  const float* be2 = (const float*)d_in[21];

  _Float16* wsh = (_Float16*)d_ws;             // needs ~4.22 MB of scratch
  convert_weights<<<4096, 256, 0, stream>>>(W0, W1, W2, wsh);

  const _Float16* W0h = wsh;
  const _Float16* W1h = wsh + 589824;
  const _Float16* W2h = wsh + 589824 + 1048576;

  size_t shmem = (size_t)(2 * ROWS * STR) * sizeof(_Float16)
               + (size_t)(2 * ROWS) * sizeof(float);   // 132,352 bytes
  (void)hipFuncSetAttribute(reinterpret_cast<const void*>(mlp_fused),
                            hipFuncAttributeMaxDynamicSharedMemorySize,
                            (int)shmem);

  int Btot = in_sizes[0] / 323;                // 65536
  int nblocks = Btot / ROWS;                   // 2048
  mlp_fused<<<nblocks, NTHREADS, shmem, stream>>>(
      x, kc, nl, opw, opb, ic, isb, ii,
      W0h, b0, W1h, b1, W2h, b2, W3, b3,
      g0, be0, g1, be1, g2, be2, (float*)d_out);

  (void)n_in; (void)out_size; (void)ws_size;
}
